// GSRNet_32719060861243
// MI455X (gfx1250) — matrix operand order repaired and checked by reading, not compiled, not run on hardware
//
#include <hip/hip_runtime.h>
#include <hip/hip_bf16.h>
#include <math.h>

// ---------------------------------------------------------------------------
// Swin block pair for MI455X (gfx1250, wave32, WMMA bf16 16x16x32, fp32 acc)
// One workgroup (256 thr = 8 waves) processes one 8x8 window end-to-end.
// All WMMA A/B operands live in LDS/global as bf16 so fragment loads are
// aligned b128 vector loads. QKV routing (q/k/vT) is tile-uniform so it
// compiles to wave-uniform branches, not per-lane EXEC masking.
// Block 2 (shifted) runs in-place on d_out (shifted partition is bijective).
// ---------------------------------------------------------------------------

typedef __bf16 bf16;
typedef __attribute__((ext_vector_type(16))) __bf16 v16bf;
typedef __attribute__((ext_vector_type(8)))  __bf16 v8bf;
typedef __attribute__((ext_vector_type(8)))  float  v8f;

#define CHN   96
#define NTOK  64
#define NHEAD 6
#define IMG   192
#define NWIN  24
#define WTOT  2304   // 4 batches * 24 * 24 windows

// LDS layout (bytes); all offsets 16B-aligned
#define OFF_X    0        // f32  [64][96]   residual / LN out          24576
#define OFF_XB   24576    // bf16 [64][96]   A-operand mirror of s_x    12288
#define OFF_Q    36864    // bf16 [6][64][32] (alias: s_aob bf16[64][96])
#define OFF_AOB  36864
#define OFF_K    61440    // bf16 [6][64][32] (alias: s_y f32[64][96])
#define OFF_Y    61440
#define OFF_V    86016    // bf16 [6][16][64]                           12288
#define OFF_SC   98304    // f32  [6][64][64] (alias: s_hb bf16[64][192])
#define OFF_HB   98304
#define OFF_SCB  196608   // bf16 [6][64][64] attn probs                49152
#define OFF_TAB  245760   // f32  [225*6] rel-pos bias table             5440
#define SMEM_BYTES (245760 + 5440)   // 251200 < 320KB/WG

// ---- fragment loaders per CDNA5 ISA VGPR layouts (wave32) ------------------
union FragU { v16bf v; v8bf h[2]; };

// A 16x32 bf16: m=lane&15; lanes<16 hold K {0..7,16..23}, lanes>=16 {8..15,24..31}
__device__ inline v16bf load_a_bf16(const bf16* base, int stride, int lane) {
  int m = lane & 15;
  int k0 = (lane < 16) ? 0 : 8;
  const bf16* r = base + m * stride + k0;
  FragU u;
  u.h[0] = *(const v8bf*)(r);        // 16B aligned
  u.h[1] = *(const v8bf*)(r + 16);   // 16B aligned
  return u.v;
}
// B 32x16 bf16: n=lane&15; lanes<16 hold K {0..7,16..23}, lanes>=16 {8..15,24..31}
__device__ inline v16bf load_b_bf16(const bf16* base, int stride, int lane) {
  int n = lane & 15;
  int k0 = (lane < 16) ? 0 : 8;
  const bf16* r = base + n * stride + k0;
  FragU u;
  u.h[0] = *(const v8bf*)(r);
  u.h[1] = *(const v8bf*)(r + 16);
  return u.v;
}

__device__ inline int region_label(int zh, int zw) {
  int hr = (zh < IMG - 8) ? 0 : ((zh < IMG - 4) ? 1 : 2);
  int wr = (zw < IMG - 8) ? 0 : ((zw < IMG - 4) ? 1 : 2);
  return hr * 3 + wr;
}

// ---- weight prepack: fp32 [K][N] -> bf16 [N][K] (B-fragment friendly) ------
__global__ void prep_w(const float* __restrict__ in, bf16* __restrict__ out,
                       int K, int N) {
  int i = blockIdx.x * blockDim.x + threadIdx.x;
  if (i >= K * N) return;
  int n = i / K, k = i % K;
  out[n * K + k] = (bf16)in[k * N + n];
}

// ---- fused window block ----------------------------------------------------
__global__ __launch_bounds__(256) void swin_block(
    const float* __restrict__ src, float* __restrict__ dst,
    const bf16* __restrict__ wqkvT, const float* __restrict__ bqkv,
    const float* __restrict__ table,
    const bf16* __restrict__ projT, const float* __restrict__ bproj,
    const float* __restrict__ g_sa, const float* __restrict__ b_sa,
    const bf16* __restrict__ w1T, const float* __restrict__ bf1,
    const bf16* __restrict__ w2T, const float* __restrict__ bf2,
    const float* __restrict__ g_ff, const float* __restrict__ b_ff,
    int shifted) {
  extern __shared__ char smem[];
  float* s_x   = (float*)(smem + OFF_X);
  bf16*  s_xb  = (bf16*) (smem + OFF_XB);
  bf16*  s_q   = (bf16*) (smem + OFF_Q);
  bf16*  s_aob = (bf16*) (smem + OFF_AOB);  // alias q (q dead after scores)
  bf16*  s_k   = (bf16*) (smem + OFF_K);
  float* s_y   = (float*)(smem + OFF_Y);    // alias k (k dead after scores)
  bf16*  s_vT  = (bf16*) (smem + OFF_V);
  float* s_sc  = (float*)(smem + OFF_SC);
  bf16*  s_hb  = (bf16*) (smem + OFF_HB);   // alias sc (sc dead after softmax)
  bf16*  s_scb = (bf16*) (smem + OFF_SCB);
  float* s_tab = (float*)(smem + OFF_TAB);

  const int tid  = threadIdx.x;
  const int lane = tid & 31;
  const int wave = tid >> 5;
  const int wid  = blockIdx.x;
  const int bidx = wid / (NWIN * NWIN);
  const int wrow = (wid % (NWIN * NWIN)) / NWIN;
  const int wcol = wid % NWIN;
  const int sh   = shifted ? 4 : 0;
  const int nlo  = lane & 15;
  const int mhal = (lane < 16) ? 0 : 8;

  // ---- load window tile [tok][ch] (f32 + bf16 mirror), table, zero q/k ----
  for (int i = tid; i < NTOK * CHN; i += 256) {
    int c = i / NTOK, t = i % NTOK;
    int hh = wrow * 8 + (t >> 3) + sh; if (hh >= IMG) hh -= IMG;
    int ww = wcol * 8 + (t & 7)  + sh; if (ww >= IMG) ww -= IMG;
    float v = src[((bidx * CHN + c) * IMG + hh) * IMG + ww];
    s_x[t * CHN + c]  = v;
    s_xb[t * CHN + c] = (bf16)v;
  }
  for (int i = tid; i < 225 * NHEAD; i += 256) s_tab[i] = table[i];
  {  // zero q+k (contiguous 49152B = 12288 floats)
    float* z = (float*)s_q;
    for (int i = tid; i < 12288; i += 256) z[i] = 0.0f;
  }
  __syncthreads();

  // ---- QKV: [64,96] @ [96,288], route into q(*0.25)/k/vT, K pad to 32 ----
  for (int t = wave; t < 4 * 18; t += 8) {
    int mt = t / 18, nt = t % 18;
    v8f acc = {};
#pragma unroll
    for (int ks = 0; ks < 96; ks += 32) {
      v16bf a = load_a_bf16(s_xb + mt * 16 * CHN + ks, CHN, lane);
      v16bf b = load_b_bf16(wqkvT + nt * 16 * CHN + ks, CHN, lane);
      acc = __builtin_amdgcn_wmma_f32_16x16x32_bf16(false, a, false, b,
                                                    (short)0, acc, false, false);
    }
    // tile-uniform routing: a 16-wide tile never straddles a 96-col boundary
    int sel  = nt / 6;   // 0=q, 1=k, 2=v   (uniform per wave iteration)
    int head = nt % 6;   // uniform per wave iteration
    int ch   = nlo;
    float bia = bqkv[nt * 16 + nlo];
    int mb = mt * 16 + mhal;
    if (sel == 0) {
#pragma unroll
      for (int e = 0; e < 8; ++e)
        s_q[(head * 64 + mb + e) * 32 + ch] = (bf16)((acc[e] + bia) * 0.25f);
    } else if (sel == 1) {
#pragma unroll
      for (int e = 0; e < 8; ++e)
        s_k[(head * 64 + mb + e) * 32 + ch] = (bf16)(acc[e] + bia);
    } else {
      v8bf vv;
#pragma unroll
      for (int e = 0; e < 8; ++e) vv[e] = (bf16)(acc[e] + bia);
      *(v8bf*)&s_vT[(head * 16 + ch) * 64 + mb] = vv;   // 16B aligned
    }
  }
  __syncthreads();

  // ---- scores: per head q[64,32p] @ k^T -> [64,64] fp32 ----
  for (int t = wave; t < NHEAD * 16; t += 8) {
    int head = t / 16, qt = (t % 16) / 4, kt = t % 4;
    v16bf a = load_a_bf16(s_q + (head * 64 + qt * 16) * 32, 32, lane);
    v16bf b = load_b_bf16(s_k + (head * 64 + kt * 16) * 32, 32, lane);
    v8f acc = {};
    acc = __builtin_amdgcn_wmma_f32_16x16x32_bf16(false, a, false, b,
                                                  (short)0, acc, false, false);
    int kcol = kt * 16 + nlo;
    int mb = qt * 16 + mhal;
#pragma unroll
    for (int e = 0; e < 8; ++e)
      s_sc[(head * 64 + mb + e) * 64 + kcol] = acc[e];
  }
  __syncthreads();

  // ---- softmax rows + rel-pos bias (+ shift mask); probs -> bf16 s_scb ----
  for (int r = tid; r < NHEAD * NTOK; r += 256) {
    int head = r / 64, q = r % 64;
    int qi = q >> 3, qj = q & 7;
    int lq = shifted ? region_label(wrow * 8 + qi, wcol * 8 + qj) : 0;
    float* row = s_sc + (head * 64 + q) * 64;
    float mx = -1e30f;
    for (int k = 0; k < 64; ++k) {
      int ki = k >> 3, kj = k & 7;
      float bia = s_tab[((qi - ki + 7) * 15 + (qj - kj + 7)) * NHEAD + head];
      if (shifted) {
        int lk = region_label(wrow * 8 + ki, wcol * 8 + kj);
        if (lk != lq) bia -= 100.0f;
      }
      float v = row[k] + bia;
      row[k] = v;
      mx = fmaxf(mx, v);
    }
    float sum = 0.0f;
    for (int k = 0; k < 64; ++k) { float e = expf(row[k] - mx); sum += e; row[k] = e; }
    float inv = 1.0f / sum;
    bf16* prow = s_scb + (head * 64 + q) * 64;
    for (int k = 0; k < 64; ++k) prow[k] = (bf16)(row[k] * inv);
  }
  __syncthreads();

  // ---- attn @ v : per head [64,64] @ [64,16] -> s_aob bf16 [64][96] ----
  for (int t = wave; t < NHEAD * 4; t += 8) {
    int head = t / 4, mt = t % 4;
    v8f acc = {};
#pragma unroll
    for (int ks = 0; ks < 64; ks += 32) {
      v16bf a = load_a_bf16(s_scb + (head * 64 + mt * 16) * 64 + ks, 64, lane);
      v16bf b = load_b_bf16(s_vT + head * 16 * 64 + ks, 64, lane);
      acc = __builtin_amdgcn_wmma_f32_16x16x32_bf16(false, a, false, b,
                                                    (short)0, acc, false, false);
    }
    int ncol = head * 16 + nlo;
    int mb = mt * 16 + mhal;
#pragma unroll
    for (int e = 0; e < 8; ++e) s_aob[(mb + e) * CHN + ncol] = (bf16)acc[e];
  }
  __syncthreads();

  // ---- proj + bias + residual -> s_y f32 ----
  for (int t = wave; t < 4 * 6; t += 8) {
    int mt = t / 6, nt = t % 6;
    v8f acc = {};
#pragma unroll
    for (int ks = 0; ks < 96; ks += 32) {
      v16bf a = load_a_bf16(s_aob + mt * 16 * CHN + ks, CHN, lane);
      v16bf b = load_b_bf16(projT + nt * 16 * CHN + ks, CHN, lane);
      acc = __builtin_amdgcn_wmma_f32_16x16x32_bf16(false, a, false, b,
                                                    (short)0, acc, false, false);
    }
    int ncol = nt * 16 + nlo;
    float bia = bproj[ncol];
    int mb = mt * 16 + mhal;
#pragma unroll
    for (int e = 0; e < 8; ++e) {
      int tok = mb + e;
      s_y[tok * CHN + ncol] = acc[e] + bia + s_x[tok * CHN + ncol];
    }
  }
  __syncthreads();

  // ---- LayerNorm (SA) : s_y -> s_x (f32) + s_xb (bf16 A mirror) ----
  if (tid < NTOK) {
    const float* row = s_y + tid * CHN;
    float m = 0.0f;
    for (int c = 0; c < CHN; ++c) m += row[c];
    m *= (1.0f / CHN);
    float v = 0.0f;
    for (int c = 0; c < CHN; ++c) { float d = row[c] - m; v += d * d; }
    float inv = 1.0f / sqrtf(v * (1.0f / CHN) + 1e-5f);
    for (int c = 0; c < CHN; ++c) {
      float o = (row[c] - m) * inv * g_sa[c] + b_sa[c];
      s_x[tid * CHN + c]  = o;
      s_xb[tid * CHN + c] = (bf16)o;
    }
  }
  __syncthreads();

  // ---- FFN1: [64,96]@[96,192] + bias, exact GELU -> s_hb bf16 ----
  for (int t = wave; t < 4 * 12; t += 8) {
    int mt = t / 12, nt = t % 12;
    v8f acc = {};
#pragma unroll
    for (int ks = 0; ks < 96; ks += 32) {
      v16bf a = load_a_bf16(s_xb + mt * 16 * CHN + ks, CHN, lane);
      v16bf b = load_b_bf16(w1T + nt * 16 * 96 + ks, 96, lane);
      acc = __builtin_amdgcn_wmma_f32_16x16x32_bf16(false, a, false, b,
                                                    (short)0, acc, false, false);
    }
    int ncol = nt * 16 + nlo;
    float bia = bf1[ncol];
    int mb = mt * 16 + mhal;
#pragma unroll
    for (int e = 0; e < 8; ++e) {
      float xv = acc[e] + bia;
      float g = 0.5f * xv * (1.0f + erff(xv * 0.70710678118f));
      s_hb[(mb + e) * 192 + ncol] = (bf16)g;
    }
  }
  __syncthreads();

  // ---- FFN2: [64,192]@[192,96] + bias + residual -> s_y f32 ----
  for (int t = wave; t < 4 * 6; t += 8) {
    int mt = t / 6, nt = t % 6;
    v8f acc = {};
#pragma unroll
    for (int ks = 0; ks < 192; ks += 32) {
      v16bf a = load_a_bf16(s_hb + mt * 16 * 192 + ks, 192, lane);
      v16bf b = load_b_bf16(w2T + nt * 16 * 192 + ks, 192, lane);
      acc = __builtin_amdgcn_wmma_f32_16x16x32_bf16(false, a, false, b,
                                                    (short)0, acc, false, false);
    }
    int ncol = nt * 16 + nlo;
    float bia = bf2[ncol];
    int mb = mt * 16 + mhal;
#pragma unroll
    for (int e = 0; e < 8; ++e) {
      int tok = mb + e;
      s_y[tok * CHN + ncol] = acc[e] + bia + s_x[tok * CHN + ncol];
    }
  }
  __syncthreads();

  // ---- LayerNorm (FF) : s_y -> s_x ----
  if (tid < NTOK) {
    const float* row = s_y + tid * CHN;
    float m = 0.0f;
    for (int c = 0; c < CHN; ++c) m += row[c];
    m *= (1.0f / CHN);
    float v = 0.0f;
    for (int c = 0; c < CHN; ++c) { float d = row[c] - m; v += d * d; }
    float inv = 1.0f / sqrtf(v * (1.0f / CHN) + 1e-5f);
    for (int c = 0; c < CHN; ++c)
      s_x[tid * CHN + c] = (row[c] - m) * inv * g_ff[c] + b_ff[c];
  }
  __syncthreads();

  // ---- store window back (same shifted coordinates => in-place safe) ----
  for (int i = tid; i < NTOK * CHN; i += 256) {
    int c = i / NTOK, t = i % NTOK;
    int hh = wrow * 8 + (t >> 3) + sh; if (hh >= IMG) hh -= IMG;
    int ww = wcol * 8 + (t & 7)  + sh; if (ww >= IMG) ww -= IMG;
    dst[((bidx * CHN + c) * IMG + hh) * IMG + ww] = s_x[t * CHN + c];
  }
}

// ---------------------------------------------------------------------------
extern "C" void kernel_launch(void* const* d_in, const int* in_sizes, int n_in,
                              void* d_out, int out_size, void* d_ws, size_t ws_size,
                              hipStream_t stream) {
  (void)in_sizes; (void)n_in; (void)out_size; (void)ws_size;
  const float* X      = (const float*)d_in[0];
  const float* qkvW1  = (const float*)d_in[1];
  const float* qkvB1  = (const float*)d_in[2];
  const float* tab1   = (const float*)d_in[3];
  const float* projW1 = (const float*)d_in[4];
  const float* projB1 = (const float*)d_in[5];
  const float* gsa1   = (const float*)d_in[6];
  const float* bsa1   = (const float*)d_in[7];
  const float* qkvW2  = (const float*)d_in[8];
  const float* qkvB2  = (const float*)d_in[9];
  const float* tab2   = (const float*)d_in[10];
  const float* projW2 = (const float*)d_in[11];
  const float* projB2 = (const float*)d_in[12];
  const float* gsa2   = (const float*)d_in[13];
  const float* bsa2   = (const float*)d_in[14];
  const float* w1a    = (const float*)d_in[15];
  const float* b1a    = (const float*)d_in[16];
  const float* w2a    = (const float*)d_in[17];
  const float* b2a    = (const float*)d_in[18];
  const float* gf1    = (const float*)d_in[19];
  const float* bff1   = (const float*)d_in[20];
  const float* w1b    = (const float*)d_in[21];
  const float* b1b    = (const float*)d_in[22];
  const float* w2b    = (const float*)d_in[23];
  const float* b2b    = (const float*)d_in[24];
  const float* gf2    = (const float*)d_in[25];
  const float* bff2   = (const float*)d_in[26];
  float* out = (float*)d_out;

  // workspace: bf16 transposed weights
  bf16* wsb = (bf16*)d_ws;
  bf16* qkvT1 = wsb;                  // 288*96
  bf16* qkvT2 = wsb + 27648;          // 288*96
  bf16* projT1 = wsb + 55296;         // 96*96
  bf16* projT2 = wsb + 64512;         // 96*96
  bf16* w1T1 = wsb + 73728;           // 192*96
  bf16* w1T2 = wsb + 92160;           // 192*96
  bf16* w2T1 = wsb + 110592;          // 96*192
  bf16* w2T2 = wsb + 129024;          // 96*192

  auto prep = [&](const float* in, bf16* o, int K, int N) {
    int tot = K * N;
    prep_w<<<(tot + 255) / 256, 256, 0, stream>>>(in, o, K, N);
  };
  prep(qkvW1, qkvT1, 96, 288);
  prep(qkvW2, qkvT2, 96, 288);
  prep(projW1, projT1, 96, 96);
  prep(projW2, projT2, 96, 96);
  prep(w1a, w1T1, 96, 192);
  prep(w1b, w1T2, 96, 192);
  prep(w2a, w2T1, 192, 96);
  prep(w2b, w2T2, 192, 96);

  // block 1 (unshifted): X -> out
  swin_block<<<WTOT, 256, SMEM_BYTES, stream>>>(
      X, out, qkvT1, qkvB1, tab1, projT1, projB1, gsa1, bsa1,
      w1T1, b1a, w2T1, b2a, gf1, bff1, 0);
  // block 2 (shifted, masked): out -> out (in-place, bijective partition)
  swin_block<<<WTOT, 256, SMEM_BYTES, stream>>>(
      out, out, qkvT2, qkvB2, tab2, projT2, projB2, gsa2, bsa2,
      w1T2, b1b, w2T2, b2b, gf2, bff2, 1);
}
